// PointNet2_11519102288168
// MI455X (gfx1250) — matrix operand order repaired and checked by reading, not compiled, not run on hardware
//
#include <hip/hip_runtime.h>
#include <cstdint>
#include <cstddef>

typedef _Float16 f16;
typedef __attribute__((ext_vector_type(16))) _Float16 v16h;
typedef __attribute__((ext_vector_type(8)))  float    v8f;

// ---------------------------------------------------------------------------
// Layout transforms
// ---------------------------------------------------------------------------

__global__ void k_transpose(const float* __restrict__ xyz, float* __restrict__ pts,
                            int B, int N) {
  int i = blockIdx.x * blockDim.x + threadIdx.x;
  int total = B * N * 3;
  if (i >= total) return;
  int c = i % 3;
  int n = (i / 3) % N;
  int b = i / (3 * N);
  pts[i] = xyz[((size_t)b * 3 + c) * N + n];
}

__global__ void k_w16(const float* __restrict__ w, f16* __restrict__ w16,
                      int D, int C, int Cpad) {
  int i = blockIdx.x * blockDim.x + threadIdx.x;
  if (i >= D * Cpad) return;
  int d = i / Cpad, c = i % Cpad;
  w16[i] = (c < C) ? (f16)w[(size_t)d * C + c] : (f16)0.f;
}

// ---------------------------------------------------------------------------
// Farthest point sampling: one workgroup per batch, dist[] resident in LDS.
// ---------------------------------------------------------------------------

__global__ void k_fps(const float* __restrict__ pts, int N, int S, int* __restrict__ out) {
  __shared__ float dist[8192];
  __shared__ float rv[256];
  __shared__ int   ri[256];
  const int b = blockIdx.x, t = threadIdx.x, T = 256;
  const float* P = pts + (size_t)b * N * 3;
  for (int n = t; n < N; n += T) dist[n] = 1e10f;
  __syncthreads();
  int far = 0;
  for (int it = 0; it < S; ++it) {
    if (t == 0) out[b * S + it] = far;
    float cx = P[far * 3 + 0], cy = P[far * 3 + 1], cz = P[far * 3 + 2];
    float best = -1.f; int bi = 0;
    for (int n = t; n < N; n += T) {
      float dx = P[n * 3 + 0] - cx, dy = P[n * 3 + 1] - cy, dz = P[n * 3 + 2] - cz;
      float d = dx * dx + dy * dy + dz * dz;
      float dm = fminf(dist[n], d);
      dist[n] = dm;
      if (dm > best) { best = dm; bi = n; }
    }
    rv[t] = best; ri[t] = bi;
    __syncthreads();
    for (int off = T / 2; off; off >>= 1) {
      if (t < off) {
        float ov = rv[t + off]; int oi = ri[t + off];
        if (ov > rv[t] || (ov == rv[t] && oi < ri[t])) { rv[t] = ov; ri[t] = oi; }
      }
      __syncthreads();
    }
    far = ri[0];
    __syncthreads();
  }
}

__global__ void k_gather3(const float* __restrict__ pts, const int* __restrict__ idx,
                          float* __restrict__ out, int N, int S, int total) {
  int i = blockIdx.x * blockDim.x + threadIdx.x;
  if (i >= total) return;                       // total = B*S
  int b = i / S;
  int g = idx[i];
  const float* p = pts + ((size_t)b * N + g) * 3;
  out[(size_t)i * 3 + 0] = p[0];
  out[(size_t)i * 3 + 1] = p[1];
  out[(size_t)i * 3 + 2] = p[2];
}

// Ordered scan == sort-by-index + take-first-nsample + pad-with-first semantics.
__global__ void k_ballq(const float* __restrict__ pts, const float* __restrict__ nxyz,
                        int* __restrict__ gidx, int total, int N, int S, int K, float r2) {
  int i = blockIdx.x * blockDim.x + threadIdx.x;  // total = B*S
  if (i >= total) return;
  int b = i / S;
  const float* q = nxyz + (size_t)i * 3;
  float qx = q[0], qy = q[1], qz = q[2];
  const float* P = pts + (size_t)b * N * 3;
  int* out = gidx + (size_t)i * K;
  int cnt = 0, first = 0; bool hasf = false;
  for (int n = 0; n < N && cnt < K; ++n) {
    float dx = P[n * 3 + 0] - qx, dy = P[n * 3 + 1] - qy, dz = P[n * 3 + 2] - qz;
    if (dx * dx + dy * dy + dz * dz <= r2) {
      if (!hasf) { first = n; hasf = true; }
      out[cnt++] = n;
    }
  }
  for (; cnt < K; ++cnt) out[cnt] = first;
}

// Grouping: centered coords (+ optional gathered point features), f16, K-padded.
__global__ void k_group(const float* __restrict__ pts, const float* __restrict__ feats,
                        const float* __restrict__ nxyz, const int* __restrict__ gidx,
                        f16* __restrict__ out, int total, int N, int S, int K,
                        int Cf, int Cpad) {
  int i = blockIdx.x * blockDim.x + threadIdx.x;  // total = B*S*K
  if (i >= total) return;
  int k = i % K;
  int bs = i / K;
  int b = bs / S;
  int g = gidx[(size_t)bs * K + k];
  const float* q = nxyz + (size_t)bs * 3;
  const float* p = pts + ((size_t)b * N + g) * 3;
  f16* o = out + (size_t)i * Cpad;
  o[0] = (f16)(p[0] - q[0]);
  o[1] = (f16)(p[1] - q[1]);
  o[2] = (f16)(p[2] - q[2]);
  int c = 3;
  if (feats) {
    const float* f = feats + ((size_t)b * N + g) * Cf;
    for (int j = 0; j < Cf; ++j) o[c++] = (f16)f[j];
  }
  for (; c < Cpad; ++c) o[c] = (f16)0.f;
}

// Layer-3 feature: raw coords (no centering) concat l2 features, padded to 288.
__global__ void k_feat3(const float* __restrict__ nxyz2, const float* __restrict__ l2feat,
                        f16* __restrict__ out, int total) {
  int m = blockIdx.x * blockDim.x + threadIdx.x;  // total = B*S2 = 2048
  if (m >= total) return;
  f16* o = out + (size_t)m * 288;
  o[0] = (f16)nxyz2[(size_t)m * 3 + 0];
  o[1] = (f16)nxyz2[(size_t)m * 3 + 1];
  o[2] = (f16)nxyz2[(size_t)m * 3 + 2];
  const float* f = l2feat + (size_t)m * 256;
  for (int j = 0; j < 256; ++j) o[3 + j] = (f16)f[j];
  for (int j = 259; j < 288; ++j) o[j] = (f16)0.f;
}

// ---------------------------------------------------------------------------
// WMMA GEMM: H[M,D] = A[M,CPAD] x W[D,CPAD]^T, f16 in, f16 out (f32 accum).
// One 16x16 tile per wave, 8 waves per block, exact grid (EXEC all-ones).
// Fragment layouts per CDNA5 ISA section 7.12.2.
// ---------------------------------------------------------------------------

template <int CPAD>
__global__ void k_gemm(const f16* __restrict__ A, const f16* __restrict__ W,
                       f16* __restrict__ H, int D) {
  const int lane  = threadIdx.x & 31;
  const int wave  = threadIdx.x >> 5;
  const int m0    = (blockIdx.x * 8 + wave) * 16;
  const int n0    = blockIdx.y * 16;
  const int khalf = (lane >> 4) * 8;
  const f16* arow = A + (size_t)(m0 + (lane & 15)) * CPAD;
  const f16* brow = W + (size_t)(n0 + (lane & 15)) * CPAD + (lane >> 4) * 8;
  v8f acc = {};
#pragma unroll
  for (int k0 = 0; k0 < CPAD; k0 += 32) {
    if (k0 + 32 < CPAD) __builtin_prefetch(arow + k0 + 32, 0, 0);
    v16h a, bf;
#pragma unroll
    for (int j = 0; j < 8; ++j) {
      int kb = k0 + ((j >= 4) ? 16 : 0) + khalf + (j & 3) * 2;
      a[2 * j]     = arow[kb];
      a[2 * j + 1] = arow[kb + 1];
    }
#pragma unroll
    for (int j = 0; j < 16; ++j) bf[j] = brow[k0 + ((j >= 8) ? 16 : 0) + (j & 7)];
    acc = __builtin_amdgcn_wmma_f32_16x16x32_f16(false, a, false, bf,
                                                 (short)0, acc, false, false);
  }
  const int mo = m0 + ((lane >> 4) << 3);
  const int n  = n0 + (lane & 15);
#pragma unroll
  for (int i = 0; i < 8; ++i)
    H[(size_t)(mo + i) * D + n] = (f16)acc[i];
}

// ---------------------------------------------------------------------------
// Deterministic per-channel sum / sumsq (BN stats over all B,S,K == all M rows)
// ---------------------------------------------------------------------------

__global__ void k_colred(const f16* __restrict__ h, float* __restrict__ cs,
                         float* __restrict__ cq, int M, int D) {
  __shared__ float ss[256], sq[256];
  int d = blockIdx.x, t = threadIdx.x;
  float s = 0.f, q = 0.f;
  for (int m = t; m < M; m += 256) {
    float v = (float)h[(size_t)m * D + d];
    s += v; q += v * v;
  }
  ss[t] = s; sq[t] = q;
  __syncthreads();
  for (int off = 128; off; off >>= 1) {
    if (t < off) { ss[t] += ss[t + off]; sq[t] += sq[t + off]; }
    __syncthreads();
  }
  if (t == 0) { cs[d] = ss[0]; cq[d] = sq[0]; }
}

__global__ void k_norm(f16* __restrict__ h, const float* __restrict__ cs,
                       const float* __restrict__ cq, const float* __restrict__ g,
                       const float* __restrict__ bb, int M, int D) {
  size_t i = (size_t)blockIdx.x * blockDim.x + threadIdx.x;
  if (i >= (size_t)M * D) return;
  int d = (int)(i % D);
  float mean = cs[d] / (float)M;
  float var  = cq[d] / (float)M - mean * mean;
  float x = (float)h[i];
  float y = g[d] * (x - mean) * rsqrtf(var + 1e-5f) + bb[d];
  h[i] = (f16)fmaxf(y, 0.f);
}

__global__ void k_maxpool(const f16* __restrict__ act, float* __restrict__ out,
                          int SB, int K, int D) {
  int i = blockIdx.x * blockDim.x + threadIdx.x;  // over SB*D
  if (i >= SB * D) return;
  int d = i % D;
  int s = i / D;
  const f16* p = act + (size_t)s * K * D + d;
  float m = -1e30f;
  for (int k = 0; k < K; ++k) m = fmaxf(m, (float)p[(size_t)k * D]);
  out[i] = m;
}

__global__ void k_final(const float* __restrict__ x, float* __restrict__ out) {
  int i = blockIdx.x * blockDim.x + threadIdx.x;
  if (i >= 4096) return;
  float v = x[i];
  out[i] = v;          // x: (B,256)
  out[4096 + i] = v;   // l3_points: (B,256,1) flattens identically
}

// ---------------------------------------------------------------------------
// Host orchestration
// ---------------------------------------------------------------------------

extern "C" void kernel_launch(void* const* d_in, const int* in_sizes, int n_in,
                              void* d_out, int out_size, void* d_ws, size_t ws_size,
                              hipStream_t stream) {
  (void)in_sizes; (void)n_in; (void)out_size; (void)ws_size;

  const int B = 16, N = 8192;
  const int S1 = 512, K1 = 32, CP1 = 32, D1a = 64, D1b = 128;
  const int S2 = 128, K2 = 64, CP2 = 160, D2a = 128, D2b = 256;
  const int CP3 = 288, D3 = 256;
  const int M1 = B * S1 * K1;   // 262144
  const int M2 = B * S2 * K2;   // 131072
  const int M3 = B * S2;        // 2048

  const float* xyz = (const float*)d_in[0];
  const float* w1a = (const float*)d_in[1];
  const float* g1a = (const float*)d_in[2];
  const float* b1a = (const float*)d_in[3];
  const float* w1b = (const float*)d_in[4];
  const float* g1b = (const float*)d_in[5];
  const float* b1b = (const float*)d_in[6];
  const float* w2a = (const float*)d_in[7];
  const float* g2a = (const float*)d_in[8];
  const float* b2a = (const float*)d_in[9];
  const float* w2b = (const float*)d_in[10];
  const float* g2b = (const float*)d_in[11];
  const float* b2b = (const float*)d_in[12];
  const float* w3  = (const float*)d_in[13];
  const float* g3  = (const float*)d_in[14];
  const float* b3  = (const float*)d_in[15];
  float* out = (float*)d_out;

  char* base = (char*)d_ws;
  size_t off = 0;
  auto alloc = [&](size_t bytes) -> char* {
    char* p = base + off;
    off += (bytes + 255) & ~(size_t)255;
    return p;
  };

  float* pts    = (float*)alloc((size_t)B * N * 3 * 4);
  int*   fps1   = (int*)  alloc((size_t)B * S1 * 4);
  float* nxyz1  = (float*)alloc((size_t)B * S1 * 3 * 4);
  int*   gidx1  = (int*)  alloc((size_t)B * S1 * K1 * 4);
  float* l1feat = (float*)alloc((size_t)B * S1 * D1b * 4);
  int*   fps2   = (int*)  alloc((size_t)B * S2 * 4);
  float* nxyz2  = (float*)alloc((size_t)B * S2 * 3 * 4);
  int*   gidx2  = (int*)  alloc((size_t)B * S2 * K2 * 4);
  float* l2feat = (float*)alloc((size_t)B * S2 * D2b * 4);
  f16*   w16_1a = (f16*)alloc((size_t)D1a * CP1 * 2);
  f16*   w16_1b = (f16*)alloc((size_t)D1b * D1a * 2);
  f16*   w16_2a = (f16*)alloc((size_t)D2a * CP2 * 2);
  f16*   w16_2b = (f16*)alloc((size_t)D2b * D2a * 2);
  f16*   w16_3  = (f16*)alloc((size_t)D3 * CP3 * 2);
  float* colsum = (float*)alloc(256 * 4);
  float* colsq  = (float*)alloc(256 * 4);
  f16*   feat3  = (f16*)alloc((size_t)M3 * CP3 * 2);
  f16*   h3     = (f16*)alloc((size_t)M3 * D3 * 2);
  float* x3     = (float*)alloc((size_t)B * D3 * 4);

  // Big scratch arena: union of layer1 (117MB) and layer2 (143MB) footprints.
  size_t l1sz = (size_t)M1 * CP1 * 2 + (size_t)M1 * D1a * 2 + (size_t)M1 * D1b * 2;
  size_t l2sz = (size_t)M2 * CP2 * 2 + (size_t)M2 * D2a * 2 + (size_t)M2 * D2b * 2;
  char* scratch = alloc((l1sz > l2sz ? l1sz : l2sz));
  f16* feat1 = (f16*)(scratch);
  f16* h1a   = (f16*)(scratch + (size_t)M1 * CP1 * 2);
  f16* h1b   = (f16*)(scratch + (size_t)M1 * CP1 * 2 + (size_t)M1 * D1a * 2);
  f16* feat2 = (f16*)(scratch);
  f16* h2a   = (f16*)(scratch + (size_t)M2 * CP2 * 2);
  f16* h2b   = (f16*)(scratch + (size_t)M2 * CP2 * 2 + (size_t)M2 * D2a * 2);

  const int T = 256;
  auto blks = [](long long n) { return dim3((unsigned)((n + 255) / 256)); };

  // Layout prep
  k_transpose<<<blks((long long)B * N * 3), T, 0, stream>>>(xyz, pts, B, N);
  k_w16<<<blks(D1a * CP1), T, 0, stream>>>(w1a, w16_1a, D1a, 3,   CP1);
  k_w16<<<blks(D1b * D1a), T, 0, stream>>>(w1b, w16_1b, D1b, 64,  D1a);
  k_w16<<<blks(D2a * CP2), T, 0, stream>>>(w2a, w16_2a, D2a, 131, CP2);
  k_w16<<<blks(D2b * D2a), T, 0, stream>>>(w2b, w16_2b, D2b, 128, D2a);
  k_w16<<<blks(D3  * CP3), T, 0, stream>>>(w3,  w16_3,  D3,  259, CP3);

  // ---------------- SA layer 1 ----------------
  k_fps<<<B, T, 0, stream>>>(pts, N, S1, fps1);
  k_gather3<<<blks(B * S1), T, 0, stream>>>(pts, fps1, nxyz1, N, S1, B * S1);
  k_ballq<<<blks(B * S1), T, 0, stream>>>(pts, nxyz1, gidx1, B * S1, N, S1, K1, 0.04f);
  k_group<<<blks(M1), T, 0, stream>>>(pts, (const float*)nullptr, nxyz1, gidx1,
                                      feat1, M1, N, S1, K1, 0, CP1);
  k_gemm<CP1><<<dim3(M1 / 128, D1a / 16), T, 0, stream>>>(feat1, w16_1a, h1a, D1a);
  k_colred<<<D1a, T, 0, stream>>>(h1a, colsum, colsq, M1, D1a);
  k_norm<<<blks((long long)M1 * D1a), T, 0, stream>>>(h1a, colsum, colsq, g1a, b1a, M1, D1a);
  k_gemm<D1a><<<dim3(M1 / 128, D1b / 16), T, 0, stream>>>(h1a, w16_1b, h1b, D1b);
  k_colred<<<D1b, T, 0, stream>>>(h1b, colsum, colsq, M1, D1b);
  k_norm<<<blks((long long)M1 * D1b), T, 0, stream>>>(h1b, colsum, colsq, g1b, b1b, M1, D1b);
  k_maxpool<<<blks((long long)B * S1 * D1b), T, 0, stream>>>(h1b, l1feat, B * S1, K1, D1b);

  // ---------------- SA layer 2 ----------------
  k_fps<<<B, T, 0, stream>>>(nxyz1, S1, S2, fps2);
  k_gather3<<<blks(B * S2), T, 0, stream>>>(nxyz1, fps2, nxyz2, S1, S2, B * S2);
  k_ballq<<<blks(B * S2), T, 0, stream>>>(nxyz1, nxyz2, gidx2, B * S2, S1, S2, K2, 0.16f);
  k_group<<<blks(M2), T, 0, stream>>>(nxyz1, l1feat, nxyz2, gidx2,
                                      feat2, M2, S1, S2, K2, D1b, CP2);
  k_gemm<CP2><<<dim3(M2 / 128, D2a / 16), T, 0, stream>>>(feat2, w16_2a, h2a, D2a);
  k_colred<<<D2a, T, 0, stream>>>(h2a, colsum, colsq, M2, D2a);
  k_norm<<<blks((long long)M2 * D2a), T, 0, stream>>>(h2a, colsum, colsq, g2a, b2a, M2, D2a);
  k_gemm<D2a><<<dim3(M2 / 128, D2b / 16), T, 0, stream>>>(h2a, w16_2b, h2b, D2b);
  k_colred<<<D2b, T, 0, stream>>>(h2b, colsum, colsq, M2, D2b);
  k_norm<<<blks((long long)M2 * D2b), T, 0, stream>>>(h2b, colsum, colsq, g2b, b2b, M2, D2b);
  k_maxpool<<<blks((long long)B * S2 * D2b), T, 0, stream>>>(h2b, l2feat, B * S2, K2, D2b);

  // ---------------- SA layer 3 (group-all) ----------------
  k_feat3<<<blks(M3), T, 0, stream>>>(nxyz2, l2feat, feat3, M3);
  k_gemm<CP3><<<dim3(M3 / 128, D3 / 16), T, 0, stream>>>(feat3, w16_3, h3, D3);
  k_colred<<<D3, T, 0, stream>>>(h3, colsum, colsq, M3, D3);
  k_norm<<<blks((long long)M3 * D3), T, 0, stream>>>(h3, colsum, colsq, g3, b3, M3, D3);
  k_maxpool<<<blks((long long)B * D3), T, 0, stream>>>(h3, x3, B, S2, D3);

  k_final<<<blks(4096), T, 0, stream>>>(x3, out);
}
